// BatchHardTripletLoss_40690520162810
// MI455X (gfx1250) — matrix operand order repaired and checked by reading, not compiled, not run on hardware
//
#include <hip/hip_runtime.h>
#include <cmath>

typedef _Float16 v16h __attribute__((ext_vector_type(16)));
typedef _Float16 v8h  __attribute__((ext_vector_type(8)));
typedef float    v8f  __attribute__((ext_vector_type(8)));
typedef unsigned int u32x4 __attribute__((ext_vector_type(4)));
typedef int          i32x8 __attribute__((ext_vector_type(8)));
typedef int          i32x4 __attribute__((ext_vector_type(4)));

#define BATCH_N   4096
#define DIM_K     512
#define MARGIN_F  0.2f
#define WAVES_PB  8              // waves per block in the main kernel
#define ROWS_PB   (WAVES_PB * 16)
#define COL_CHUNKS 8
#define TILES_PER_CHUNK (BATCH_N / 16 / COL_CHUNKS)   // 32
#define B_PITCH_H 520            // padded LDS pitch in halfs (1040 B)

#if defined(__HIP_DEVICE_COMPILE__) && __has_builtin(__builtin_amdgcn_tensor_load_to_lds)
#define USE_TDM 1
#else
#define USE_TDM 0
#endif

// ---- order-preserving float <-> uint encoding (for atomic min/max) --------
__device__ __forceinline__ unsigned f2ord(float f) {
  unsigned u = __float_as_uint(f);
  return (u & 0x80000000u) ? ~u : (u | 0x80000000u);
}
__device__ __forceinline__ float ord2f(unsigned u) {
  return __uint_as_float((u & 0x80000000u) ? (u & 0x7fffffffu) : ~u);
}

// ---------------------------------------------------------------------------
// Kernel 0: init per-anchor encoded min/max arrays.
// ---------------------------------------------------------------------------
__global__ void bh_init(unsigned* __restrict__ minPosEnc,
                        unsigned* __restrict__ maxNegEnc) {
  const int i = blockIdx.x * 256 + threadIdx.x;
  minPosEnc[i] = f2ord(1e30f);
  maxNegEnc[i] = f2ord(-1e30f);
}

// ---------------------------------------------------------------------------
// Kernel 1: L2-normalize each 512-float row, emit f16 row-major copy.
// ---------------------------------------------------------------------------
__global__ void __launch_bounds__(128)
bh_normalize_rows(const float* __restrict__ emb, _Float16* __restrict__ e16) {
  const int row = blockIdx.x;
  const int tid = threadIdx.x;
  const float* src = emb + (size_t)row * DIM_K;

  float v[4];
  float s = 0.0f;
#pragma unroll
  for (int i = 0; i < 4; ++i) {
    v[i] = src[tid + i * 128];
    s += v[i] * v[i];
  }
#pragma unroll
  for (int off = 16; off > 0; off >>= 1)
    s += __shfl_xor(s, off, 32);

  __shared__ float red[4];
  if ((tid & 31) == 0) red[tid >> 5] = s;
  __syncthreads();
  const float tot = red[0] + red[1] + red[2] + red[3];
  const float inv = 1.0f / fmaxf(sqrtf(tot), 1e-12f);

  _Float16* dst = e16 + (size_t)row * DIM_K;
#pragma unroll
  for (int i = 0; i < 4; ++i)
    dst[tid + i * 128] = (_Float16)(v[i] * inv);
}

// ---------------------------------------------------------------------------
// TDM: issue a tensor_load_to_lds of one contiguous B tile (16 rows x 1024 B)
// with LDS padding of 16 B after every 1024 B -> padded pitch 1040 B.
// D# built per CDNA5 ISA (group0/group1 bit layout).
// ---------------------------------------------------------------------------
#if USE_TDM
__device__ __forceinline__ void tdm_load_tile(const _Float16* gsrc,
                                              unsigned ldsByteOff) {
  const unsigned long long ga = (unsigned long long)(uintptr_t)gsrc;
  u32x4 g0;
  g0[0] = 1u;                                        // count=1, no gather
  g0[1] = ldsByteOff;                                // lds_addr
  g0[2] = (unsigned)(ga & 0xffffffffu);              // global_addr[31:0]
  g0[3] = (unsigned)((ga >> 32) & 0x01ffffffu)       // global_addr[56:32]
        | (2u << 30);                                // type=2 (image)
  i32x8 g1;
  // data_size=3 (8B), pad_enable, pad_interval=7 (256 dw), pad_amount=3 (4 dw)
  g1[0] = (int)((3u << 16) | (1u << 20) | (7u << 22) | (3u << 25));
  g1[1] = (int)(2048u << 16);        // tensor_dim0[15:0]=2048 @bits63:48
  g1[2] = (int)(1u << 16);           // tensor_dim0 hi=0; tensor_dim1=1
  g1[3] = (int)(2048u << 16);        // tensor_dim1 hi=0; tile_dim0=2048
  g1[4] = 0;                         // tile_dim1=0, tile_dim2=0 (1-D tile)
  g1[5] = 2048;                      // tensor_dim0_stride lo
  g1[6] = 0;
  g1[7] = 0;
  i32x4 g2 = {0, 0, 0, 0};
  i32x4 g3 = {0, 0, 0, 0};
#if __clang_major__ >= 23
  i32x8 g4 = {0, 0, 0, 0, 0, 0, 0, 0};
  __builtin_amdgcn_tensor_load_to_lds(g0, g1, g2, g3, g4, 0);
#else
  __builtin_amdgcn_tensor_load_to_lds(g0, g1, g2, g3, 0);
#endif
}
#endif

// ---------------------------------------------------------------------------
// Kernel 2: fused Gram-matrix + batch-hard mining.
// Grid (32, 8): x = 128-anchor row group, y = 512-candidate column chunk.
// Block = 8 waves; wave w owns row tile (16 anchors) with its A panel in
// VGPRs (16 x v16h). B tiles are TDM-streamed into double-buffered LDS and
// shared by all 8 waves; two accumulators break the v_wmma RAW chain, and
// the B fragment DS loads are software-pipelined at depth 2.
// ---------------------------------------------------------------------------
__global__ void __launch_bounds__(256)
bh_batch_hard_wmma(const _Float16* __restrict__ e16,
                   const int* __restrict__ labels,
                   unsigned* __restrict__ minPosEnc,
                   unsigned* __restrict__ maxNegEnc) {
  __shared__ _Float16 ldsB[2][16 * B_PITCH_H];   // 2 x 16.25 KB tiles

  const int tid  = threadIdx.x;
  const int wave = tid >> 5;
  const int lane = tid & 31;
  const int hsel = lane >> 4;     // half-wave select
  const int nl   = lane & 15;     // row-in-tile for A loads / N for B,C

  const int rowTile  = blockIdx.x * WAVES_PB + wave;
  const int tileBase = blockIdx.y * TILES_PER_CHUNK;

  // --- preload this wave's A panel (16 anchors x 512) into registers ------
  // A layout: lane row = L&15; lanes<16 take K {0..7,16..23}+k0,
  // lanes>=16 take K {8..15,24..31}+k0  -> two 16B chunks per K step.
  const _Float16* arow = e16 + (size_t)(rowTile * 16 + nl) * DIM_K;
  v16h areg[16];
#pragma unroll
  for (int k = 0; k < 16; ++k) {
    const v8h a0 = *(const v8h*)(arow + 32 * k + hsel * 8);
    const v8h a1 = *(const v8h*)(arow + 32 * k + 16 + hsel * 8);
    areg[k] = __builtin_shufflevector(a0, a1, 0, 1, 2, 3, 4, 5, 6, 7, 8, 9,
                                      10, 11, 12, 13, 14, 15);
  }

  // C-tile layout: VGPR r, lane L -> M = r + 8*hsel, N = L&15.
  int mlab[8], mrow[8];
#pragma unroll
  for (int r = 0; r < 8; ++r) {
    mrow[r] = rowTile * 16 + hsel * 8 + r;
    mlab[r] = labels[mrow[r]];
  }

  float minPos[8], maxNeg[8];
#pragma unroll
  for (int r = 0; r < 8; ++r) { minPos[r] = 1e30f; maxNeg[r] = -1e30f; }

#if USE_TDM
  if (wave == 0)   // one TDM issue per block; EXEC is ignored by TDM
    tdm_load_tile(e16 + (size_t)(tileBase * 16) * DIM_K,
                  (unsigned)(uintptr_t)&ldsB[0][0]);
#endif

  for (int t = 0; t < TILES_PER_CHUNK; ++t) {
    const int ct = tileBase + t;
    const int ng = ct * 16 + nl;            // global candidate index
    const int nlab = labels[ng];

#if USE_TDM
    if (wave == 0) {
      if (t + 1 < TILES_PER_CHUNK) {
        tdm_load_tile(e16 + (size_t)((ct + 1) * 16) * DIM_K,
                      (unsigned)(uintptr_t)&ldsB[(t + 1) & 1][0]);
        __builtin_amdgcn_s_wait_tensorcnt((short)1);  // tile t resident
      } else {
        __builtin_amdgcn_s_wait_tensorcnt((short)0);
      }
    }
    __syncthreads();   // broadcast "tile t ready"
#else
    // Fallback: cooperative copy of the 16 KB tile with padded pitch.
    {
      const _Float16* src = e16 + (size_t)(ct * 16) * DIM_K;
      _Float16* dst = &ldsB[t & 1][0];
#pragma unroll
      for (int i = tid; i < 1024; i += 256) {        // 1024 x 16 B
        const int row = i >> 6, chunk = i & 63;
        *(uint4*)(dst + row * B_PITCH_H + chunk * 8) =
            *(const uint4*)(src + row * DIM_K + chunk * 8);
      }
      __syncthreads();
    }
#endif

    // B layout (32x16 f16): lane holds K = k0+8*hsel..+7 and k0+16+8*hsel..+7
    // of column nl, i.e. two 16-byte chunks of LDS row nl.
    const _Float16* brow = &ldsB[t & 1][nl * B_PITCH_H + hsel * 8];

    v8f c0 = {0.f, 0.f, 0.f, 0.f, 0.f, 0.f, 0.f, 0.f};
    v8f c1 = {0.f, 0.f, 0.f, 0.f, 0.f, 0.f, 0.f, 0.f};

    // Depth-2 software pipeline on the B fragment loads: while wmma(k)
    // executes, the DS loads for steps k+1 and k+2 are in flight.
    v8h bl[3], bh[3];
    bl[0] = *(const v8h*)(brow + 0);
    bh[0] = *(const v8h*)(brow + 16);
    bl[1] = *(const v8h*)(brow + 32);
    bh[1] = *(const v8h*)(brow + 48);
#pragma unroll
    for (int k = 0; k < 16; ++k) {
      const int cur = k % 3;
      if (k + 2 < 16) {
        const int nxt = (k + 2) % 3;
        bl[nxt] = *(const v8h*)(brow + 32 * (k + 2));
        bh[nxt] = *(const v8h*)(brow + 32 * (k + 2) + 16);
      }
      const v16h b = __builtin_shufflevector(bl[cur], bh[cur], 0, 1, 2, 3, 4,
                                             5, 6, 7, 8, 9, 10, 11, 12, 13,
                                             14, 15);
      if (k & 1)
        c1 = __builtin_amdgcn_wmma_f32_16x16x32_f16(false, areg[k], false, b,
                                                    (short)0, c1, false,
                                                    false);
      else
        c0 = __builtin_amdgcn_wmma_f32_16x16x32_f16(false, areg[k], false, b,
                                                    (short)0, c0, false,
                                                    false);
    }

    // Masked hardest mining on the 16x16 dot tile.
#pragma unroll
    for (int r = 0; r < 8; ++r) {
      const float d = c0[r] + c1[r];
      if (mrow[r] != ng) {
        if (mlab[r] == nlab) minPos[r] = fminf(minPos[r], d);
        else                 maxNeg[r] = fmaxf(maxNeg[r], d);
      }
    }
    __syncthreads();   // all waves done reading tile t before it is reused
  }

  // Reduce across the 16 lanes of each half-wave.
#pragma unroll
  for (int r = 0; r < 8; ++r) {
#pragma unroll
    for (int off = 8; off > 0; off >>= 1) {
      minPos[r] = fminf(minPos[r], __shfl_xor(minPos[r], off, 32));
      maxNeg[r] = fmaxf(maxNeg[r], __shfl_xor(maxNeg[r], off, 32));
    }
  }

  // Lane 0 holds M=0..7, lane 16 holds M=8..15; merge chunks via atomics.
  if (nl == 0) {
#pragma unroll
    for (int r = 0; r < 8; ++r) {
      atomicMin(&minPosEnc[mrow[r]], f2ord(minPos[r]));
      atomicMax(&maxNegEnc[mrow[r]], f2ord(maxNeg[r]));
    }
  }
}

// ---------------------------------------------------------------------------
// Kernel 3: decode per-anchor results, hinge, mean -> out[0]. Single block.
// ---------------------------------------------------------------------------
__global__ void __launch_bounds__(256)
bh_finalize(const unsigned* __restrict__ minPosEnc,
            const unsigned* __restrict__ maxNegEnc,
            float* __restrict__ out) {
  float lsum = 0.0f, lcnt = 0.0f;
  for (int i = threadIdx.x; i < BATCH_N; i += 256) {
    const float mp = ord2f(minPosEnc[i]);   // min positive dot
    const float mn = ord2f(maxNegEnc[i]);   // max negative dot
    if ((mp < 1e29f) && (mn > -1e29f)) {
      // hp - hn = (1 - mp) - (1 - mn) = mn - mp
      lsum += fmaxf(0.0f, mn - mp + MARGIN_F);
      lcnt += 1.0f;
    }
  }
#pragma unroll
  for (int off = 16; off > 0; off >>= 1) {
    lsum += __shfl_xor(lsum, off, 32);
    lcnt += __shfl_xor(lcnt, off, 32);
  }
  __shared__ float ss[8], sc[8];
  if ((threadIdx.x & 31) == 0) {
    ss[threadIdx.x >> 5] = lsum;
    sc[threadIdx.x >> 5] = lcnt;
  }
  __syncthreads();
  if (threadIdx.x == 0) {
    float s = 0.0f, c = 0.0f;
#pragma unroll
    for (int w = 0; w < 8; ++w) { s += ss[w]; c += sc[w]; }
    out[0] = s / fmaxf(c, 1.0f);
  }
}

// ---------------------------------------------------------------------------
extern "C" void kernel_launch(void* const* d_in, const int* in_sizes, int n_in,
                              void* d_out, int out_size, void* d_ws,
                              size_t ws_size, hipStream_t stream) {
  const float* emb    = (const float*)d_in[0];
  const int*   labels = (const int*)d_in[1];

  _Float16* e16 = (_Float16*)d_ws;                                   // 4 MB
  unsigned* minPosEnc =
      (unsigned*)((char*)d_ws + (size_t)BATCH_N * DIM_K * sizeof(_Float16));
  unsigned* maxNegEnc = minPosEnc + BATCH_N;

  hipLaunchKernelGGL(bh_init, dim3(BATCH_N / 256), dim3(256), 0, stream,
                     minPosEnc, maxNegEnc);
  hipLaunchKernelGGL(bh_normalize_rows, dim3(BATCH_N), dim3(128), 0, stream,
                     emb, e16);
  hipLaunchKernelGGL(bh_batch_hard_wmma,
                     dim3(BATCH_N / ROWS_PB, COL_CHUNKS), dim3(256), 0,
                     stream, e16, labels, minPosEnc, maxNegEnc);
  hipLaunchKernelGGL(bh_finalize, dim3(1), dim3(256), 0, stream, minPosEnc,
                     maxNegEnc, (float*)d_out);
}
